// MultiHeadSelfAttention_84146999263472
// MI455X (gfx1250) — matrix operand order repaired and checked by reading, not compiled, not run on hardware
//
#include <hip/hip_runtime.h>

// ---------------------------------------------------------------------------
// MHSA for MI455X (gfx1250): f16 WMMA (f32 accum) everywhere.
//   k1: QKV = x @ w_qkv + b  -> Q,K [b,h,n,hd] f16; V transposed [b,h,hd,n] f16
//   k2: flash attention; TDM (tensor_load_to_lds) double-buffers K/V tiles into
//       LDS shared by the 4 waves of a block; 8 wmma per 32-key block per wave.
//   k3: out = ctx @ w_proj + b -> f32
// ---------------------------------------------------------------------------

typedef __attribute__((ext_vector_type(16))) _Float16 v16h;
typedef __attribute__((ext_vector_type(8)))  float    v8f;
typedef __attribute__((ext_vector_type(4)))  unsigned int u32x4;
typedef __attribute__((ext_vector_type(8)))  int          i32x8;
typedef __attribute__((ext_vector_type(4)))  int          i32x4;

#define B_  2
#define N_  2048
#define D_  1024
#define H_  16
#define HD_ 64

// A-matrix (16x32 f16) K index for vgpr-pair i, per ISA 7.12.2 table.
__device__ __forceinline__ int frag_kA(int i, int lane) {
  return 2 * i + ((i >= 4) ? 8 : 0) + ((lane >= 16) ? 8 : 0);
}
// B-matrix (32x16 f16) K index (same table order as A).
__device__ __forceinline__ int frag_kB(int i, int lane) {
  return 2 * i + ((i >= 4) ? 8 : 0) + ((lane >= 16) ? 8 : 0);
}

__device__ __forceinline__ v8f wmma16(v16h a, v16h b, v8f c) {
  return __builtin_amdgcn_wmma_f32_16x16x32_f16(
      /*neg_a=*/false, a, /*neg_b=*/false, b,
      /*c_mod=*/(short)0, c, /*reuse_a=*/false, /*reuse_b=*/false);
}

// LDS byte offset of a __shared__ object (flat LDS aperture addr[31:0] == offset).
__device__ __forceinline__ unsigned lds_off(const void* p) {
  return (unsigned)(uintptr_t)p;
}

// ---------------------------------------------------------------------------
// Tensor Data Mover: 2D tile load Global -> LDS.
//   data_size = 8 bytes; dim0_units/stride_units in 8B units; rows = dim1.
// Descriptor layout per CDNA5 ISA ch.8 (Group0 128b, Group1 256b; 2/3 zeroed).
// ---------------------------------------------------------------------------
__device__ __forceinline__ void tdm_load_2d(const void* gptr, unsigned lds_addr,
                                            unsigned dim0_units, unsigned rows,
                                            unsigned stride_units) {
  unsigned long long ga = (unsigned long long)(uintptr_t)gptr;
  u32x4 g0;
  g0[0] = 1u;                                        // count=1 (valid user D#)
  g0[1] = lds_addr;                                  // lds_addr [63:32]
  g0[2] = (unsigned)(ga & 0xffffffffu);              // global_addr lo
  g0[3] = (unsigned)((ga >> 32) & 0x1ffffffu) | (2u << 30);  // addr hi | type=2
  i32x8 g1;
  g1[0] = (3 << 16);                                 // data_size = 8B, no flags
  g1[1] = (int)((dim0_units & 0xffffu) << 16);       // tensor_dim0[15:0]
  g1[2] = (int)(((dim0_units >> 16) & 0xffffu) |     // tensor_dim0[31:16]
                ((rows & 0xffffu) << 16));           // tensor_dim1[15:0]
  g1[3] = (int)(((rows >> 16) & 0xffffu) |           // tensor_dim1[31:16]
                ((dim0_units & 0xffffu) << 16));     // tile_dim0
  g1[4] = (int)(rows & 0xffffu);                     // tile_dim1 (tile_dim2=0)
  g1[5] = (int)stride_units;                         // tensor_dim0_stride lo32
  g1[6] = 0;                                         // stride hi | dim1_stride
  g1[7] = 0;
  i32x4 z4 = {};
#if __clang_major__ >= 23
  i32x8 z8 = {};
  __builtin_amdgcn_tensor_load_to_lds(g0, g1, z4, z4, z8, 0);
#else
  __builtin_amdgcn_tensor_load_to_lds(g0, g1, z4, z4, 0);
#endif
}

// ---------------------------------------------------------------------------
// Tiled GEMM: block = 128 thr (4 waves), tile = 16 rows x 256 cols,
// each wave owns 16x64 of output. K advanced 32 at a time through LDS.
// MODE 0: A=f32 x, epilogue scatters f16 Q/K/V(T).  MODE 1: A=f16 ctx, f32 out.
// ---------------------------------------------------------------------------
template <int MODE>
__global__ __launch_bounds__(128) void gemm_tiled(
    const void* __restrict__ Ap, const float* __restrict__ W,
    const float* __restrict__ bias,
    _Float16* __restrict__ qOut, _Float16* __restrict__ kOut,
    _Float16* __restrict__ vOut, float* __restrict__ fOut,
    int K, int NC) {
  __shared__ _Float16 xs[16][32];       // A tile (row-major, k contiguous)
  __shared__ _Float16 wt[256][34];      // B slab transposed [col][k], padded

  const int tid = threadIdx.x, wv = tid >> 5, lane = tid & 31;
  const int rowBase = blockIdx.x * 16;
  const int colBase = blockIdx.y * 256;
  const int rA = lane & 15;
  const int grp8 = (lane >= 16) ? 8 : 0;

  v8f zero = {};
  v8f acc[4];
#pragma unroll
  for (int t = 0; t < 4; ++t) acc[t] = zero;

  const float*    Af = (const float*)Ap;
  const _Float16* Ah = (const _Float16*)Ap;

  for (int kk = 0; kk < K; kk += 32) {
    __syncthreads();
    // stage A tile 16x32
    for (int e = tid; e < 16 * 32; e += 128) {
      int r = e >> 5, c = e & 31;
      float v = (MODE == 0) ? Af[(size_t)(rowBase + r) * K + kk + c]
                            : (float)Ah[(size_t)(rowBase + r) * K + kk + c];
      xs[r][c] = (_Float16)v;
    }
    // stage W slab 32x256, transposed into [col][k]; coalesced over cols
    for (int e = tid; e < 256 * 32; e += 128) {
      int kq = e >> 8, c = e & 255;
      wt[c][kq] = (_Float16)W[(size_t)(kk + kq) * NC + colBase + c];
    }
    if (kk + 32 < K)  // global_prefetch_b8 of next slab
      __builtin_prefetch(&W[(size_t)(kk + 32) * NC + colBase + (tid & 255)], 0, 1);
    __syncthreads();

    v16h a;
#pragma unroll
    for (int i = 0; i < 8; ++i) {
      int k0 = frag_kA(i, lane);
      a[2 * i]     = xs[rA][k0];
      a[2 * i + 1] = xs[rA][k0 + 1];
    }
#pragma unroll
    for (int t = 0; t < 4; ++t) {
      v16h b;
      int cn = wv * 64 + t * 16 + rA;
#pragma unroll
      for (int i = 0; i < 8; ++i) {
        int k0 = frag_kB(i, lane);
        b[2 * i]     = wt[cn][k0];
        b[2 * i + 1] = wt[cn][k0 + 1];
      }
      acc[t] = wmma16(a, b, acc[t]);
    }
  }

  // epilogue
#pragma unroll
  for (int t = 0; t < 4; ++t) {
    int c = colBase + wv * 64 + t * 16 + rA;
    float bval = bias[c];
#pragma unroll
    for (int r = 0; r < 8; ++r) {
      int row = rowBase + r + grp8;
      float val = acc[t][r] + bval;
      if (MODE == 0) {
        int bb = row >> 11, n = row & (N_ - 1);
        int which = c >> 10, dc = c & (D_ - 1);
        int h = dc >> 6, hd = dc & 63;
        size_t bh = (size_t)bb * H_ + h;
        _Float16 hv = (_Float16)val;
        if (which == 0)      qOut[(bh * N_ + n) * HD_ + hd] = hv;
        else if (which == 1) kOut[(bh * N_ + n) * HD_ + hd] = hv;
        else                 vOut[(bh * HD_ + hd) * N_ + n] = hv;  // transposed
      } else {
        fOut[(size_t)row * NC + c] = val;
      }
    }
  }
}

// ---------------------------------------------------------------------------
// Flash attention: block = 4 waves, ALL sharing one (b,h) (128 q-tiles per bh,
// 4 consecutive tiles per block). Wave 0 drives the TDM: K block (32x64 f16,
// contiguous) and V block (64 rows x 32 keys, stride N_) are DMA'd into LDS,
// double-buffered, overlapping the previous block's 8 WMMAs + online softmax.
// ---------------------------------------------------------------------------
__global__ __launch_bounds__(128) void attn_flash(
    const _Float16* __restrict__ Q, const _Float16* __restrict__ Kw,
    const _Float16* __restrict__ Vt, _Float16* __restrict__ Ctx) {
  __shared__ _Float16 kbuf[2][32][64];  // [buf][key][hd]   8 KB
  __shared__ _Float16 vbuf[2][64][32];  // [buf][hd][key]   8 KB
  __shared__ _Float16 pbuf[4][16][32];  // per-wave P relayout scratch 4 KB

  const int tid = threadIdx.x, wv = tid >> 5, lane = tid & 31;
  const int gw = blockIdx.x * 4 + wv;       // 0..4095
  const int qtile = gw & 127;
  const int bh = gw >> 7;                   // uniform across the block
  const int b = bh >> 4, h = bh & 15;
  const int rA = lane & 15;
  const int grp8 = (lane >= 16) ? 8 : 0;

  const _Float16* qb = Q  + (size_t)bh * N_ * HD_;
  const _Float16* kb = Kw + (size_t)bh * N_ * HD_;   // [n][hd]
  const _Float16* vb = Vt + (size_t)bh * HD_ * N_;   // [hd][n]

  // Prologue: DMA first K/V blocks into buffer 0 (wave 0 only; EXEC ignored).
  if (wv == 0) {
    tdm_load_2d(kb,           lds_off(&kbuf[0][0][0]), 16, 32, 16);   // 4KB contig
    tdm_load_2d(vb,           lds_off(&vbuf[0][0][0]), 8, 64, N_ / 4);// strided
  }

  // Q fragments with 1/sqrt(HD)=0.125 folded in (exact in f16)
  const _Float16 sc = (_Float16)0.125f;
  const size_t qoff = (size_t)(qtile * 16 + rA) * HD_;
  v16h aQ0, aQ1;
#pragma unroll
  for (int i = 0; i < 8; ++i) {
    int k0 = frag_kA(i, lane);
    aQ0[2 * i]     = qb[qoff + k0] * sc;
    aQ0[2 * i + 1] = qb[qoff + k0 + 1] * sc;
    aQ1[2 * i]     = qb[qoff + k0 + 32] * sc;
    aQ1[2 * i + 1] = qb[qoff + k0 + 33] * sc;
  }

  v8f zero = {};
  v8f o[4];
  float m[8], l[8];
#pragma unroll
  for (int t = 0; t < 4; ++t) o[t] = zero;
#pragma unroll
  for (int r = 0; r < 8; ++r) { m[r] = -1e30f; l[r] = 0.0f; }

  for (int jb = 0; jb < N_; jb += 32) {
    const int ping = (jb >> 5) & 1;
    // Current buffers ready; publish to all waves.
    if (wv == 0) __builtin_amdgcn_s_wait_tensorcnt(0);
    __syncthreads();
    // Kick next block's DMA into the other buffer (overlaps compute below).
    if (wv == 0 && jb + 32 < N_) {
      tdm_load_2d(kb + (size_t)(jb + 32) * HD_,
                  lds_off(&kbuf[ping ^ 1][0][0]), 16, 32, 16);
      tdm_load_2d(vb + (jb + 32),
                  lds_off(&vbuf[ping ^ 1][0][0]), 8, 64, N_ / 4);
    }

    // ---- S = Q K^T : two 16x16 tiles (keys jb..+15 and jb+16..+31) ----
    v8f s0 = zero, s1 = zero;
    {
      v16h bk;
#pragma unroll
      for (int i = 0; i < 8; ++i) {
        int k0 = frag_kB(i, lane);                 // hd index
        bk[2 * i] = kbuf[ping][rA][k0]; bk[2 * i + 1] = kbuf[ping][rA][k0 + 1];
      }
      s0 = wmma16(aQ0, bk, s0);
#pragma unroll
      for (int i = 0; i < 8; ++i) {
        int k0 = frag_kB(i, lane);
        bk[2 * i] = kbuf[ping][rA][k0 + 32]; bk[2 * i + 1] = kbuf[ping][rA][k0 + 33];
      }
      s0 = wmma16(aQ1, bk, s0);
#pragma unroll
      for (int i = 0; i < 8; ++i) {
        int k0 = frag_kB(i, lane);
        bk[2 * i] = kbuf[ping][16 + rA][k0]; bk[2 * i + 1] = kbuf[ping][16 + rA][k0 + 1];
      }
      s1 = wmma16(aQ0, bk, s1);
#pragma unroll
      for (int i = 0; i < 8; ++i) {
        int k0 = frag_kB(i, lane);
        bk[2 * i] = kbuf[ping][16 + rA][k0 + 32]; bk[2 * i + 1] = kbuf[ping][16 + rA][k0 + 33];
      }
      s1 = wmma16(aQ1, bk, s1);
    }

    // ---- online softmax over the 16-column lane group ----
#pragma unroll
    for (int r = 0; r < 8; ++r) {
      float mx = fmaxf(s0[r], s1[r]);
      mx = fmaxf(mx, __shfl_xor(mx, 1, 32));
      mx = fmaxf(mx, __shfl_xor(mx, 2, 32));
      mx = fmaxf(mx, __shfl_xor(mx, 4, 32));
      mx = fmaxf(mx, __shfl_xor(mx, 8, 32));
      float mn = fmaxf(m[r], mx);
      float alpha = __expf(m[r] - mn);
      float p0 = __expf(s0[r] - mn);
      float p1 = __expf(s1[r] - mn);
      float rs = p0 + p1;
      rs += __shfl_xor(rs, 1, 32);
      rs += __shfl_xor(rs, 2, 32);
      rs += __shfl_xor(rs, 4, 32);
      rs += __shfl_xor(rs, 8, 32);
      l[r] = l[r] * alpha + rs;
      m[r] = mn;
#pragma unroll
      for (int t = 0; t < 4; ++t) o[t][r] *= alpha;
      pbuf[wv][r + grp8][rA]      = (_Float16)p0;   // C-layout store
      pbuf[wv][r + grp8][rA + 16] = (_Float16)p1;
    }

    // ---- P: C-layout -> A-layout (wave-private LDS; same-wave DS is in-order)
    v16h aP;
#pragma unroll
    for (int i = 0; i < 8; ++i) {
      int k0 = frag_kA(i, lane);
      aP[2 * i]     = pbuf[wv][rA][k0];
      aP[2 * i + 1] = pbuf[wv][rA][k0 + 1];
    }

    // ---- O += P V : vbuf is [hd][key], key pairs contiguous ----
#pragma unroll
    for (int t = 0; t < 4; ++t) {
      v16h bv;
      int hd = t * 16 + rA;
#pragma unroll
      for (int i = 0; i < 8; ++i) {
        int k0 = frag_kB(i, lane);                 // key-in-block
        bv[2 * i] = vbuf[ping][hd][k0]; bv[2 * i + 1] = vbuf[ping][hd][k0 + 1];
      }
      o[t] = wmma16(aP, bv, o[t]);
    }
  }

  // ---- normalize and store context f16: [b, n, h*64+hd] ----
#pragma unroll
  for (int r = 0; r < 8; ++r) {
    int row = qtile * 16 + r + grp8;
    float inv = 1.0f / l[r];
    size_t base = ((size_t)b * N_ + row) * D_ + h * HD_;
#pragma unroll
    for (int t = 0; t < 4; ++t) {
      int hd = t * 16 + rA;
      Ctx[base + hd] = (_Float16)(o[t][r] * inv);
    }
  }
}

// ---------------------------------------------------------------------------
extern "C" void kernel_launch(void* const* d_in, const int* in_sizes, int n_in,
                              void* d_out, int out_size, void* d_ws, size_t ws_size,
                              hipStream_t stream) {
  const float* x      = (const float*)d_in[0];
  const float* w_qkv  = (const float*)d_in[1];
  const float* b_qkv  = (const float*)d_in[2];
  const float* w_proj = (const float*)d_in[3];
  const float* b_proj = (const float*)d_in[4];
  float* out = (float*)d_out;

  const size_t elems = (size_t)B_ * N_ * D_;     // 4 Mi f16 per tensor
  _Float16* qws = (_Float16*)d_ws;               // [b,h,n,hd]
  _Float16* kws = qws + elems;                   // [b,h,n,hd]
  _Float16* vws = kws + elems;                   // [b,h,hd,n] (transposed)
  _Float16* ctx = vws + elems;                   // [b,n,d]

  // QKV projection: 4096x3072 output, K=1024
  gemm_tiled<0><<<dim3((B_ * N_) / 16, (3 * D_) / 256), 128, 0, stream>>>(
      x, w_qkv, b_qkv, qws, kws, vws, nullptr, D_, 3 * D_);

  // Flash attention: 4096 q-tiles, 4 waves/block (one bh per block)
  attn_flash<<<(B_ * H_ * (N_ / 16)) / 4, 128, 0, stream>>>(qws, kws, vws, ctx);

  // Output projection: 4096x1024, K=1024
  gemm_tiled<1><<<dim3((B_ * N_) / 16, D_ / 256), 128, 0, stream>>>(
      ctx, w_proj, b_proj, nullptr, nullptr, nullptr, out, D_, D_);
}
